// spatialAttentionGCN_48816598286772
// MI455X (gfx1250) — compile-verified, matrix operand order checked by reading, not run on hardware
//
#include <hip/hip_runtime.h>

// ---------------- CDNA5 (gfx1250) fused spatial-attention GCN ----------------
// x:(8,1024,32,64) f32, adj:(1024,1024) f32, theta:(64,64) f32 -> out:(8,1024,32,64) f32
// Flash-style fused attention per (b,t) slice; bf16 WMMA (v_wmma_f32_16x16x32_bf16),
// f32 accumulate; masked value-accumulation with UNMASKED softmax denominator.
// exp2-domain online softmax; native bf16 converts; global_prefetch for next tile.

typedef __attribute__((ext_vector_type(16))) __bf16 v16bf;
typedef __attribute__((ext_vector_type(8)))  __bf16 v8bf;
typedef __attribute__((ext_vector_type(4)))  __bf16 v4bf;
typedef __attribute__((ext_vector_type(8)))  float  v8f;
typedef __attribute__((ext_vector_type(4)))  float  v4f;

#define NB 8
#define NN 1024
#define NT 32
#define NF 64
#define BLOCK_M 128
#define BLOCK_N 128
#define NWAVES  8
#define QSTR (NF + 8)        // 72 bf16 -> 144B row stride (16B aligned)
#define TSTR (BLOCK_N + 8)   // 136 bf16 -> 272B row stride (16B aligned)

// native convert: backend emits packed bf16 cvt ops on gfx1250
__device__ __forceinline__ __bf16 f2bf(float f) { return (__bf16)f; }

// 16-bit A/B fragment (16x32 tile): lane l holds row (l&15); with h=(l>>4),
// elements e=0..7  come from K = koff + 8h + e        (16 contiguous bytes)
// elements e=8..15 come from K = koff + 8h + 16 + e-8 (16 contiguous bytes)
__device__ __forceinline__ v16bf load_frag(const __bf16* p0) {
  v8bf lo = *(const v8bf*)(p0);
  v8bf hi = *(const v8bf*)(p0 + 16);
  return __builtin_shufflevector(lo, hi, 0,1,2,3,4,5,6,7,8,9,10,11,12,13,14,15);
}

__device__ __forceinline__ v8f wmma_bf16(v16bf a, v16bf b, v8f c) {
  return __builtin_amdgcn_wmma_f32_16x16x32_bf16(false, a, false, b, (short)0, c,
                                                 false, false);
}

__device__ __forceinline__ float redmax16(float v) {
#pragma unroll
  for (int off = 8; off; off >>= 1) v = fmaxf(v, __shfl_xor(v, off, 32));
  return v;
}
__device__ __forceinline__ float redsum16(float v) {
#pragma unroll
  for (int off = 8; off; off >>= 1) v += __shfl_xor(v, off, 32);
  return v;
}

__global__ __launch_bounds__(256, 1)
void sagcn_fused_kernel(const float* __restrict__ x,
                        const float* __restrict__ adj,
                        const float* __restrict__ theta,
                        float* __restrict__ out) {
  __shared__ __attribute__((aligned(16))) __bf16 Qs[BLOCK_M * QSTR];   // 18 KB
  __shared__ __attribute__((aligned(16))) __bf16 Ks[BLOCK_N * QSTR];   // 18 KB
  __shared__ __attribute__((aligned(16))) __bf16 KsT[NF * TSTR];       // 17 KB
  __shared__ __attribute__((aligned(16))) __bf16 Wt[NF * NF];          //  8 KB
  __shared__ __attribute__((aligned(16))) __bf16 Ps[NWAVES * 16 * BLOCK_N]; // 32 KB

  const int tid  = threadIdx.x;
  const int lane = tid & 31;
  const int wave = tid >> 5;
  const int h    = (lane >> 4) & 1;
  const int ln   = lane & 15;

  const int bid  = blockIdx.x;
  const int bt   = bid >> 3;            // (b*T + t)
  const int rblk = bid & 7;
  const int b    = bt / NT;
  const int t    = bt % NT;
  const int r0   = rblk * BLOCK_M;

  const size_t slice_base = (size_t)b * NN * NT * NF + (size_t)t * NF;
  const int    row_stride = NT * NF;    // 2048 floats between n and n+1

  // ---- stage theta (64x64 f32 -> bf16, row-major [f][k]) ----
  for (int i = tid; i < NF * NF; i += 256) Wt[i] = f2bf(theta[i]);

  // ---- stage Q tile: rows r0..r0+127 of this (b,t) slice ----
  {
    const int cc = (tid & 15) * 4;      // 16 threads/row, float4 each
    const int rr = tid >> 4;            // 16 rows/pass
#pragma unroll
    for (int pass = 0; pass < 8; ++pass) {
      const int r = rr + pass * 16;
      v4f val = *(const v4f*)(x + slice_base + (size_t)(r0 + r) * row_stride + cc);
      v4bf bv = { f2bf(val.x), f2bf(val.y), f2bf(val.z), f2bf(val.w) };
      *(v4bf*)&Qs[r * QSTR + cc] = bv;
    }
  }
  __syncthreads();

  // per-wave Q A-fragments (rows wave*16 .. +16, K = 0..63)
  const int wrow = wave * 16;
  const v16bf qf0 = load_frag(&Qs[(wrow + ln) * QSTR + 0  + 8 * h]);
  const v16bf qf1 = load_frag(&Qs[(wrow + ln) * QSTR + 32 + 8 * h]);

  float mstate[8], lstate[8];
#pragma unroll
  for (int v = 0; v < 8; ++v) { mstate[v] = -1e30f; lstate[v] = 0.0f; }
  v8f Oc[4];
#pragma unroll
  for (int ft = 0; ft < 4; ++ft) Oc[ft] = (v8f){0, 0, 0, 0, 0, 0, 0, 0};

  const float inv_sqrt_f = 0.125f;                    // 1/sqrt(64)
  const float scl = 0.125f * 1.44269504088896f;       // inv_sqrt_f * log2(e)
  __bf16* Pw = &Ps[wave * 16 * BLOCK_N];
  const int grow = r0 + wrow;           // global row base for this wave

  for (int m0 = 0; m0 < NN; m0 += BLOCK_N) {
    __syncthreads();                    // all waves done reading previous K tile
    // ---- stage K/V tile rows m0..m0+127: row-major (Ks) and f-major (KsT) ----
    {
      const int cc = (tid & 15) * 4;
      const int rr = tid >> 4;
#pragma unroll
      for (int pass = 0; pass < 8; ++pass) {
        const int r = rr + pass * 16;
        const float* src = x + slice_base + (size_t)(m0 + r) * row_stride + cc;
        v4f val = *(const v4f*)src;
        if (m0 + BLOCK_N < NN)          // prefetch next K tile into caches
          __builtin_prefetch(src + (size_t)BLOCK_N * row_stride, 0, 3);
        __bf16 b0 = f2bf(val.x), b1 = f2bf(val.y), b2 = f2bf(val.z), b3 = f2bf(val.w);
        *(v4bf*)&Ks[r * QSTR + cc] = (v4bf){ b0, b1, b2, b3 };
        KsT[(cc + 0) * TSTR + r] = b0;
        KsT[(cc + 1) * TSTR + r] = b1;
        KsT[(cc + 2) * TSTR + r] = b2;
        KsT[(cc + 3) * TSTR + r] = b3;
      }
    }
    __syncthreads();

    // prefetch next adjacency segment for this wave's rows (L2-resident 4MB array)
    if (m0 + BLOCK_N < NN) {
#pragma unroll
      for (int v = 0; v < 8; ++v)
        __builtin_prefetch(&adj[(size_t)(grow + v + 8 * h) * NN + (m0 + BLOCK_N) + ln * 8],
                           0, 3);
    }

    // ---- S = (Q K^T) * inv_sqrt_f * log2e : 8 tiles of 16x16, f32 accum ----
    v8f S[8];
#pragma unroll
    for (int jt = 0; jt < 8; ++jt) {
      v16bf kf0 = load_frag(&Ks[(jt * 16 + ln) * QSTR + 0  + 8 * h]);
      v16bf kf1 = load_frag(&Ks[(jt * 16 + ln) * QSTR + 32 + 8 * h]);
      v8f acc = (v8f){0, 0, 0, 0, 0, 0, 0, 0};
      acc = wmma_bf16(qf0, kf0, acc);
      acc = wmma_bf16(qf1, kf1, acc);
      S[jt] = acc;
    }

    // ---- online softmax stats in exp2 domain (rows = v + 8h in C-layout) ----
    float tmax[8];
#pragma unroll
    for (int v = 0; v < 8; ++v) tmax[v] = -1e30f;
#pragma unroll
    for (int jt = 0; jt < 8; ++jt)
#pragma unroll
      for (int v = 0; v < 8; ++v) {
        S[jt][v] *= scl;
        tmax[v] = fmaxf(tmax[v], S[jt][v]);
      }
#pragma unroll
    for (int v = 0; v < 8; ++v) tmax[v] = redmax16(tmax[v]);

    float alpha[8];
#pragma unroll
    for (int v = 0; v < 8; ++v) {
      float mn = fmaxf(mstate[v], tmax[v]);
      alpha[v] = exp2f(mstate[v] - mn);
      mstate[v] = mn;
    }

    // ---- p = exp2(s - m); denominator UNMASKED; masked p -> Ps (bf16) ----
    float psum[8];
#pragma unroll
    for (int v = 0; v < 8; ++v) psum[v] = 0.0f;
#pragma unroll
    for (int jt = 0; jt < 8; ++jt) {
      const int gcol = m0 + jt * 16 + ln;
#pragma unroll
      for (int v = 0; v < 8; ++v) {
        float p = exp2f(S[jt][v] - mstate[v]);
        psum[v] += p;
        float a = adj[(size_t)(grow + v + 8 * h) * NN + gcol];
        Pw[(v + 8 * h) * BLOCK_N + jt * 16 + ln] = f2bf(p * a);
      }
    }
#pragma unroll
    for (int v = 0; v < 8; ++v) {
      psum[v] = redsum16(psum[v]);
      lstate[v] = lstate[v] * alpha[v] + psum[v];
    }
    // rescale running O (same row mapping as S)
#pragma unroll
    for (int ft = 0; ft < 4; ++ft)
#pragma unroll
      for (int v = 0; v < 8; ++v) Oc[ft][v] *= alpha[v];

    // ---- O += P @ V  (K2 = 128 in four 32-wide steps) ----
#pragma unroll
    for (int ks = 0; ks < 4; ++ks) {
      v16bf pf = load_frag(&Pw[ln * BLOCK_N + ks * 32 + 8 * h]);
#pragma unroll
      for (int ft = 0; ft < 4; ++ft) {
        v16bf vf = load_frag(&KsT[(ft * 16 + ln) * TSTR + ks * 32 + 8 * h]);
        Oc[ft] = wmma_bf16(pf, vf, Oc[ft]);
      }
    }
  }

  // ---- epilogue: normalize, Y = relu(O @ W^T), scatter to (B,N,T,F) ----
#pragma unroll
  for (int v = 0; v < 8; ++v) {
    float inv = inv_sqrt_f / lstate[v];
#pragma unroll
    for (int ft = 0; ft < 4; ++ft) Oc[ft][v] *= inv;
  }
  // stage O as 16x64 bf16 A-operand (wave-private; same-wave DS ops are in-order)
#pragma unroll
  for (int ft = 0; ft < 4; ++ft)
#pragma unroll
    for (int v = 0; v < 8; ++v)
      Pw[(v + 8 * h) * NF + ft * 16 + ln] = f2bf(Oc[ft][v]);

  v16bf of0 = load_frag(&Pw[ln * NF + 0  + 8 * h]);
  v16bf of1 = load_frag(&Pw[ln * NF + 32 + 8 * h]);
#pragma unroll
  for (int ft = 0; ft < 4; ++ft) {
    v16bf wf0 = load_frag(&Wt[(ft * 16 + ln) * NF + 0  + 8 * h]);
    v16bf wf1 = load_frag(&Wt[(ft * 16 + ln) * NF + 32 + 8 * h]);
    v8f y = (v8f){0, 0, 0, 0, 0, 0, 0, 0};
    y = wmma_bf16(of0, wf0, y);
    y = wmma_bf16(of1, wf1, y);
#pragma unroll
    for (int v = 0; v < 8; ++v) {
      const int n = grow + v + 8 * h;
      out[slice_base + (size_t)n * row_stride + ft * 16 + ln] = fmaxf(y[v], 0.0f);
    }
  }
}

extern "C" void kernel_launch(void* const* d_in, const int* in_sizes, int n_in,
                              void* d_out, int out_size, void* d_ws, size_t ws_size,
                              hipStream_t stream) {
  (void)in_sizes; (void)n_in; (void)out_size; (void)d_ws; (void)ws_size;
  const float* x     = (const float*)d_in[0];
  const float* adj   = (const float*)d_in[1];
  const float* theta = (const float*)d_in[2];
  float* out = (float*)d_out;

  const int grid = (NB * NT) * (NN / BLOCK_M);   // 256 * 8 = 2048 workgroups
  sagcn_fused_kernel<<<grid, 256, 0, stream>>>(x, adj, theta, out);
}
